// EncounterAutoencoder_32736240730117
// MI455X (gfx1250) — matrix operand order repaired and checked by reading, not compiled, not run on hardware
//
#include <hip/hip_runtime.h>

// ---------------------------------------------------------------------------
// 20-layer LSTM (H=128, E=768, B=64, T=512) for MI455X (gfx1250, wave32).
//
// Kernel 1: big WMMA GEMM  PRE0[t*64+b, n] = x[b,t,:]·Wih0[n,:] + bih0+bhh0 (f16)
// Kernel 2: persistent wavefront pipeline, 20 workgroups (1 per layer / WGP).
//   - fused [Wih;Whh] weights resident in LDS as f16 (270KB, padded rows)
//   - per step: y_in(t) tile (16KB f16) staged into LDS via
//     global_load_async_to_lds_b128 (+ s_wait_asynccnt), then one
//     64x512x256 GEMM as v_wmma_f32_16x16x32_f16 tiles
//   - gates i/f/g/o land per-wave => cell state c lives entirely in VGPRs
//   - layer l spins on layer l-1's progress flag (release fence + flag /
//     acquire fence before consuming)
// ---------------------------------------------------------------------------

typedef __attribute__((ext_vector_type(16))) _Float16 v16h;
typedef __attribute__((ext_vector_type(8)))  _Float16 v8h;
typedef __attribute__((ext_vector_type(8)))  float    v8f;
typedef __attribute__((ext_vector_type(4)))  float    v4f;

namespace {
constexpr int Hd  = 128;   // hidden
constexpr int Ed  = 768;   // embedding
constexpr int NL  = 20;    // layers
constexpr int Bb  = 64;    // batch
constexpr int Tt  = 512;   // seq len
constexpr int Gd  = 512;   // 4*H
constexpr int XS  = 272;   // padded LDS row stride for [y|h] operand (halves)
constexpr int WS1 = 264;   // padded LDS W row stride, K=256 (halves)
constexpr int WS0 = 136;   // padded LDS W row stride, K=128 (halves)
}

__device__ __forceinline__ v8f wmma_f16(v16h a, v16h b, v8f c) {
  return __builtin_amdgcn_wmma_f32_16x16x32_f16(false, a, false, b,
                                                (short)0, c, false, false);
}

__device__ __forceinline__ v16h pack16(v8h lo, v8h hi) {
  v16h a;
#pragma unroll
  for (int i = 0; i < 8; ++i) { a[i] = lo[i]; a[8 + i] = hi[i]; }
  return a;
}

__device__ __forceinline__ float fast_sigmoid(float x) {
  float e = __builtin_amdgcn_exp2f(-1.44269504088896340736f * x);
  return __builtin_amdgcn_rcpf(1.0f + e);
}
__device__ __forceinline__ float fast_tanh(float x) {
  float e = __builtin_amdgcn_exp2f(-2.88539008177792681472f * x);
  return __builtin_amdgcn_rcpf(1.0f + e) * 2.0f - 1.0f;
}

// ---------------------------------------------------------------------------
// Kernel 1: PRE0 = xs @ Wih0^T + (bih0+bhh0), output f16 [T*B, 512]
// ---------------------------------------------------------------------------
__global__ __launch_bounds__(256) void lstm_pre0_gemm(
    const float* __restrict__ x, const float* __restrict__ Wih0,
    const float* __restrict__ bih0, const float* __restrict__ bhh0,
    _Float16* __restrict__ pre0) {
  const int lane = threadIdx.x & 31;
  const int wave = threadIdx.x >> 5;
  const int tile = blockIdx.x * 8 + wave;   // 65536 tiles
  const int mt = tile >> 5;                 // 0..2047
  const int nt = tile & 31;                 // 0..31
  const int n  = (nt << 4) + (lane & 15);
  const int m  = (mt << 4) + (lane & 15);
  const int kb = (lane >> 4) << 3;          // 0 or 8

  const float* arow = x + ((size_t)(m & (Bb - 1)) * Tt + (m >> 6)) * Ed;
  const float* brow = Wih0 + (size_t)n * Ed;
  const float bias  = bih0[n] + bhh0[n];

  v8f acc;
#pragma unroll
  for (int r = 0; r < 8; ++r) acc[r] = bias;

  for (int k0 = 0; k0 < Ed; k0 += 32) {
    const float* pa = arow + k0 + kb;
    v4f x0 = *(const v4f*)(pa);
    v4f x1 = *(const v4f*)(pa + 4);
    v4f x2 = *(const v4f*)(pa + 16);
    v4f x3 = *(const v4f*)(pa + 20);
    const float* pb = brow + k0 + kb;
    v4f w0 = *(const v4f*)(pb);
    v4f w1 = *(const v4f*)(pb + 4);
    v4f w2 = *(const v4f*)(pb + 16);
    v4f w3 = *(const v4f*)(pb + 20);
    v16h a, b;
#pragma unroll
    for (int i = 0; i < 4; ++i) {
      a[i] = (_Float16)x0[i]; a[4 + i] = (_Float16)x1[i];
      a[8 + i] = (_Float16)x2[i]; a[12 + i] = (_Float16)x3[i];
      b[i] = (_Float16)w0[i]; b[4 + i] = (_Float16)w1[i];
      b[8 + i] = (_Float16)w2[i]; b[12 + i] = (_Float16)w3[i];
    }
    acc = wmma_f16(a, b, acc);
  }
#pragma unroll
  for (int r = 0; r < 8; ++r) {
    int mr = (mt << 4) + ((lane >> 4) << 3) + r;
    pre0[(size_t)mr * Gd + n] = (_Float16)acc[r];
  }
}

// ---------------------------------------------------------------------------
// Per-layer recurrence body.  Operand buffer xlds[64][XS] holds
// cols 0..127 = y_in(t) (async-staged; layers>=1), cols 128..255 = h(t-1).
// Layer 0 keeps h in cols 0..127 (K=128 only, PRE0 provides the input term).
// ---------------------------------------------------------------------------
template <bool IS0>
__device__ __forceinline__ void lstm_layer_run(
    int layer, int tid,
    const _Float16* __restrict__ pre0,
    const _Float16* __restrict__ yin, _Float16* __restrict__ yout,
    const float* __restrict__ bih, const float* __restrict__ bhh,
    float* __restrict__ out, int* __restrict__ prog,
    _Float16* Wlds, _Float16* xlds) {
  constexpr int KT    = IS0 ? 4 : 8;   // K-tiles of 32
  constexpr int WS    = IS0 ? WS0 : WS1;
  constexpr int HBASE = IS0 ? 0 : 128; // h column offset inside xlds
  const int lane = tid & 31;
  const int wave = tid >> 5;

  float bg[4];
  if (!IS0) {
#pragma unroll
    for (int g = 0; g < 4; ++g) {
      int n = (((g << 3) + wave) << 4) + (lane & 15);
      bg[g] = bih[(size_t)(layer - 1) * Gd + n] + bhh[(size_t)(layer - 1) * Gd + n];
    }
  }

  v8f cst[4];
#pragma unroll
  for (int mt = 0; mt < 4; ++mt)
#pragma unroll
    for (int r = 0; r < 8; ++r) cst[mt][r] = 0.0f;

  for (int t = 0; t < Tt; ++t) {
    // ---- wait for producer (layer-1) to publish y(t), then async-stage it ----
    if (!IS0) {
      if (tid == 0) {
        while (__atomic_load_n(&prog[layer - 1], __ATOMIC_RELAXED) <= t)
          __builtin_amdgcn_s_sleep(1);
      }
      __syncthreads();
      __builtin_amdgcn_fence(__ATOMIC_ACQUIRE, "agent");  // inv caches for fresh y

      // stage 64x128 f16 tile into xlds cols 0..127 via async DMA-to-LDS:
      // 1024 x 16B chunks, 4 per thread.
      const _Float16* ysrc = yin + (size_t)t * (Bb * Hd);
#pragma unroll
      for (int i = 0; i < 4; ++i) {
        const int c  = tid + (i << 8);          // 0..1023
        const int m  = c >> 4;                  // row 0..63
        const int ko = (c & 15) << 3;           // col 0..120 (halves)
        unsigned long long ga =
            (unsigned long long)(uintptr_t)(ysrc + (size_t)m * Hd + ko);
        unsigned int la =
            (unsigned int)(uintptr_t)(xlds + (size_t)m * XS + ko);
        asm volatile("global_load_async_to_lds_b128 %0, %1, off"
                     :: "v"(la), "v"(ga) : "memory");
      }
    }

    // ---- accumulator init: bias (layers>=1) or PRE0 fragment (layer 0) ----
    v8f acc[4][4];
    if (IS0) {
      const _Float16* pt = pre0 + (size_t)t * (Bb * Gd);
#pragma unroll
      for (int mt = 0; mt < 4; ++mt) {
#pragma unroll
        for (int g = 0; g < 4; ++g) {
          const int col = (((g << 3) + wave) << 4) + (lane & 15);
#pragma unroll
          for (int r = 0; r < 8; ++r) {
            const int mm = (mt << 4) + ((lane >> 4) << 3) + r;
            acc[mt][g][r] = (float)pt[(size_t)mm * Gd + col];
          }
        }
      }
    } else {
#pragma unroll
      for (int mt = 0; mt < 4; ++mt)
#pragma unroll
        for (int g = 0; g < 4; ++g)
#pragma unroll
          for (int r = 0; r < 8; ++r) acc[mt][g][r] = bg[g];
    }

    if (!IS0) {
      asm volatile("s_wait_asynccnt 0" ::: "memory");  // my chunks in LDS
      __syncthreads();                                 // everyone's chunks
    }

    // ---- G = [y_in | h] @ Wcat^T, K-outer (each B fragment reused x4 mt) ----
#pragma unroll
    for (int kt = 0; kt < KT; ++kt) {
      v16h a4[4];
#pragma unroll
      for (int mt = 0; mt < 4; ++mt) {
        const _Float16* p = xlds + ((mt << 4) + (lane & 15)) * XS +
                            (kt << 5) + ((lane >> 4) << 3);
        a4[mt] = pack16(*(const v8h*)p, *(const v8h*)(p + 16));
      }
#pragma unroll
      for (int g = 0; g < 4; ++g) {
        const int n = (((g << 3) + wave) << 4) + (lane & 15);
        const _Float16* bp = Wlds + (size_t)n * WS + (kt << 5) + ((lane >> 4) << 3);
        v16h bfr = pack16(*(const v8h*)bp, *(const v8h*)(bp + 16));
#pragma unroll
        for (int mt = 0; mt < 4; ++mt)
          acc[mt][g] = wmma_f16(a4[mt], bfr, acc[mt][g]);
      }
    }

    __syncthreads();   // all A-fragment reads done before overwriting h

    // ---- gates + cell update, all in registers ----
#pragma unroll
    for (int mt = 0; mt < 4; ++mt) {
#pragma unroll
      for (int r = 0; r < 8; ++r) {
        float gi = fast_sigmoid(acc[mt][0][r]);
        float gf = fast_sigmoid(acc[mt][1][r]);
        float gg = fast_tanh   (acc[mt][2][r]);
        float go = fast_sigmoid(acc[mt][3][r]);
        float c  = gf * cst[mt][r] + gi * gg;
        cst[mt][r] = c;
        float h  = go * fast_tanh(c);
        const int mm  = (mt << 4) + ((lane >> 4) << 3) + r;  // batch row
        const int col = (wave << 4) + (lane & 15);           // h column
        xlds[mm * XS + HBASE + col] = (_Float16)h;
        if (yout) yout[((size_t)t * Bb + mm) * Hd + col] = (_Float16)h;
        else if (t == Tt - 1) out[mm * Hd + col] = h;        // last layer
      }
    }

    // ---- publish y(t) (release) ----
    if (yout) {
      __builtin_amdgcn_fence(__ATOMIC_RELEASE, "agent");
      __syncthreads();
      if (tid == 0) __atomic_store_n(&prog[layer], t + 1, __ATOMIC_RELAXED);
    } else {
      __syncthreads();
    }
  }
}

// ---------------------------------------------------------------------------
// Kernel 2: persistent pipeline, grid = 20 blocks x 256 threads, dynamic LDS.
// ---------------------------------------------------------------------------
__global__ __launch_bounds__(256, 1) void lstm_pipeline_kernel(
    const _Float16* __restrict__ pre0,
    const float* __restrict__ Wih, const float* __restrict__ Whh,
    const float* __restrict__ Whh0,
    const float* __restrict__ bih, const float* __restrict__ bhh,
    _Float16* __restrict__ ybuf, float* __restrict__ out,
    int* __restrict__ prog) {
  extern __shared__ _Float16 smem[];
  const int layer = blockIdx.x;
  const int tid   = threadIdx.x;
  const bool is0  = (layer == 0);
  const int WS  = is0 ? WS0 : WS1;
  const int kcs = is0 ? 4 : 5;                 // log2(KC/8)
  const int nch = (is0 ? Gd * Hd : Gd * 2 * Hd) >> 3;  // 8-half chunks
  _Float16* Wlds = smem;                       // [512][WS] f16, padded rows
  _Float16* xlds = smem + (size_t)Gd * WS;     // [64][XS]  f16

  // load + convert fused weights into LDS, 8 halves per thread-iteration
  for (int c = tid; c < nch; c += 256) {
    const int nr = c >> kcs;
    const int k8 = (c & ((1 << kcs) - 1)) << 3;
    const float* src;
    if (is0)          src = Whh0 + (size_t)nr * Hd + k8;
    else if (k8 < Hd) src = Wih + ((size_t)(layer - 1) * Gd + nr) * Hd + k8;
    else              src = Whh + ((size_t)(layer - 1) * Gd + nr) * Hd + (k8 - Hd);
    v4f w0 = *(const v4f*)(src);
    v4f w1 = *(const v4f*)(src + 4);
    v8h o;
#pragma unroll
    for (int i = 0; i < 4; ++i) {
      o[i] = (_Float16)w0[i];
      o[4 + i] = (_Float16)w1[i];
    }
    *(v8h*)(Wlds + (size_t)nr * WS + k8) = o;
  }
  for (int idx = tid; idx < Bb * XS; idx += 256) xlds[idx] = (_Float16)0.0f;
  __syncthreads();

  const _Float16* yin = is0 ? nullptr : (ybuf + (size_t)(layer - 1) * Tt * Bb * Hd);
  _Float16* yout = (layer < NL - 1) ? (ybuf + (size_t)layer * Tt * Bb * Hd) : nullptr;

  if (is0) lstm_layer_run<true >(layer, tid, pre0, yin, yout, bih, bhh, out, prog, Wlds, xlds);
  else     lstm_layer_run<false>(layer, tid, pre0, yin, yout, bih, bhh, out, prog, Wlds, xlds);
}

__global__ void lstm_init_flags(int* prog) {
  if (threadIdx.x < 32) prog[threadIdx.x] = 0;
}

// ---------------------------------------------------------------------------
extern "C" void kernel_launch(void* const* d_in, const int* in_sizes, int n_in,
                              void* d_out, int out_size, void* d_ws, size_t ws_size,
                              hipStream_t stream) {
  (void)in_sizes; (void)n_in; (void)out_size; (void)ws_size;
  const float* x    = (const float*)d_in[0];
  const float* Wih0 = (const float*)d_in[1];
  const float* Whh0 = (const float*)d_in[2];
  const float* bih0 = (const float*)d_in[3];
  const float* bhh0 = (const float*)d_in[4];
  const float* Wih  = (const float*)d_in[5];
  const float* Whh  = (const float*)d_in[6];
  const float* bih  = (const float*)d_in[7];
  const float* bhh  = (const float*)d_in[8];
  float* out = (float*)d_out;

  // workspace layout: [flags 256B][PRE0 f16 32MB][ybuf f16 19*8MB]
  char* ws = (char*)d_ws;
  int* prog = (int*)ws;
  _Float16* pre0 = (_Float16*)(ws + 256);
  _Float16* ybuf = (_Float16*)(ws + 256 + (size_t)Tt * Bb * Gd * sizeof(_Float16));

  hipLaunchKernelGGL(lstm_init_flags, dim3(1), dim3(32), 0, stream, prog);

  hipLaunchKernelGGL(lstm_pre0_gemm, dim3((Tt * Bb / 16) * (Gd / 16) / 8),
                     dim3(256), 0, stream, x, Wih0, bih0, bhh0, pre0);

  const size_t smem_bytes = (size_t)Gd * WS1 * sizeof(_Float16) +
                            (size_t)Bb * XS * sizeof(_Float16);   // 305152 < 320KB
  hipFuncSetAttribute((const void*)lstm_pipeline_kernel,
                      hipFuncAttributeMaxDynamicSharedMemorySize,
                      (int)smem_bytes);
  hipLaunchKernelGGL(lstm_pipeline_kernel, dim3(NL), dim3(256), smem_bytes,
                     stream, pre0, Wih, Whh, Whh0, bih, bhh, ybuf, out, prog);
}
